// SlideGraphArch_37666863186371
// MI455X (gfx1250) — matrix operand order repaired and checked by reading, not compiled, not run on hardware
//
#include <hip/hip_runtime.h>

#define NN 100000
#define NE 1600000
#define D_IN 1024
#define D_H 64
#define D_OUT 4

typedef __attribute__((ext_vector_type(16))) __bf16 v16bf;
typedef __attribute__((ext_vector_type(8)))  float  v8f;
typedef int v4i_vs __attribute__((vector_size(16)));

// ---------------------------------------------------------------------------
// Async global->LDS copy of 16 bytes per lane (CDNA5 GLOBAL_LOAD_ASYNC_TO_LDS_B128,
// tracked by ASYNCcnt), with inline-asm fallback.
// Builtin signature on this toolchain: (AS1 int4*, AS3 int4*, imm offset, imm cpol)
// ---------------------------------------------------------------------------
__device__ __forceinline__ void async_copy16(const void* g, void* l) {
#if __has_builtin(__builtin_amdgcn_global_load_async_to_lds_b128)
  typedef __attribute__((address_space(1))) v4i_vs* gptr_t;
  typedef __attribute__((address_space(3))) v4i_vs* lptr_t;
  __builtin_amdgcn_global_load_async_to_lds_b128(
      (gptr_t)const_cast<void*>(g), (lptr_t)l, 0, 0);
#else
  unsigned lofs = (unsigned)(size_t)(__attribute__((address_space(3))) char*)l;
  asm volatile("global_load_async_to_lds_b128 %0, %1, off"
               :: "v"(lofs), "v"(g) : "memory");
#endif
}

template <int N>
__device__ __forceinline__ void wait_async() {
#if __has_builtin(__builtin_amdgcn_s_wait_asynccnt)
  __builtin_amdgcn_s_wait_asynccnt(N);
#else
  asm volatile("s_wait_asynccnt %0" :: "i"(N) : "memory");
#endif
}

// ---------------------------------------------------------------------------
// Pack a (K x 64) f32 weight matrix into per-lane WMMA B-fragment layout,
// split into bf16 hi + lo parts.
// B-fragment layout for V_WMMA_F32_16X16X32_BF16 (B is 32x16):
//   lanes 0-15 hold column n, K={0..7,16..23}; lanes 16-31 hold K={8..15,24..31}.
// Packed as [ks][nt][hi/lo][lane][16 bf16] blocks of 512 bf16 (1 KB).
// ---------------------------------------------------------------------------
__global__ void k_pack_w(const float* __restrict__ W, __bf16* __restrict__ packed, int K) {
  int idx = blockIdx.x * blockDim.x + threadIdx.x;
  if (idx >= K * D_H) return;
  int k = idx >> 6, n = idx & 63;
  float f = W[idx];
  __bf16 hi = (__bf16)f;
  __bf16 lo = (__bf16)(f - (float)hi);
  int ks   = k >> 5;
  int kin  = k & 31;
  int lane = (((kin >> 3) & 1) << 4) + (n & 15);
  int i    = (kin & 7) + ((kin >> 4) << 3);
  int nt   = n >> 4;
  int base = (ks * 8 + nt * 2) * 512 + lane * 16 + i;
  packed[base]       = hi;
  packed[base + 512] = lo;
}

// ---------------------------------------------------------------------------
// WMMA GEMM: out[N x 64] = (A (+ A2)) [N x K] @ B[K x 64] + bias
// A in f32, split on the fly to bf16 hi+lo -> 3 bf16 WMMAs per accumulate.
// Packed B streamed to LDS with double-buffered async copies shared by the
// block's 8 waves.
// ---------------------------------------------------------------------------
__global__ __launch_bounds__(256) void k_wmma_gemm(
    const float* __restrict__ A, const float* __restrict__ A2,
    const __bf16* __restrict__ pB, const float* __restrict__ bias,
    float* __restrict__ out, int N, int K) {
  __shared__ __bf16 lb[2][8 * 512];   // 2 x 8 KB double buffer
  const int tid  = threadIdx.x;
  const int wave = tid >> 5;
  const int lane = tid & 31;
  const int half = lane >> 4;         // 0 or 1
  const int l15  = lane & 15;
  const int m0   = (blockIdx.x * 8 + wave) * 16;

  int row = m0 + l15;
  if (row >= N) row = N - 1;          // clamp; stores are guarded
  const size_t arow = (size_t)row * K;

  v8f acc[4] = {v8f{}, v8f{}, v8f{}, v8f{}};

  const char* gB = (const char*)pB;
  const int o0 = tid * 16;            // byte offsets within an 8 KB chunk
  const int o1 = 4096 + tid * 16;

  // prologue: prefetch chunk 0 into buffer 0
  async_copy16(gB + o0, (char*)lb[0] + o0);
  async_copy16(gB + o1, (char*)lb[0] + o1);

  const int nks = K >> 5;
  for (int ks = 0; ks < nks; ++ks) {
    const int cur = ks & 1;
    __syncthreads();                  // all waves done with lb[cur^1]

    // A fragment (independent of LDS): lane<16 -> K {0..7,16..23};
    // lane>=16 -> K {8..15,24..31}
    const int kofs = ks * 32 + half * 8;
    float af[16];
    {
      const float* ap = A + arow + kofs;
      float4 f0 = *(const float4*)(ap);
      float4 f1 = *(const float4*)(ap + 4);
      float4 f2 = *(const float4*)(ap + 16);
      float4 f3 = *(const float4*)(ap + 20);
      af[0]=f0.x;  af[1]=f0.y;  af[2]=f0.z;  af[3]=f0.w;
      af[4]=f1.x;  af[5]=f1.y;  af[6]=f1.z;  af[7]=f1.w;
      af[8]=f2.x;  af[9]=f2.y;  af[10]=f2.z; af[11]=f2.w;
      af[12]=f3.x; af[13]=f3.y; af[14]=f3.z; af[15]=f3.w;
      if (A2) {   // GIN: A = h + agg
        const float* gp = A2 + arow + kofs;
        float4 g0 = *(const float4*)(gp);
        float4 g1 = *(const float4*)(gp + 4);
        float4 g2 = *(const float4*)(gp + 16);
        float4 g3 = *(const float4*)(gp + 20);
        af[0]+=g0.x;  af[1]+=g0.y;  af[2]+=g0.z;  af[3]+=g0.w;
        af[4]+=g1.x;  af[5]+=g1.y;  af[6]+=g1.z;  af[7]+=g1.w;
        af[8]+=g2.x;  af[9]+=g2.y;  af[10]+=g2.z; af[11]+=g2.w;
        af[12]+=g3.x; af[13]+=g3.y; af[14]+=g3.z; af[15]+=g3.w;
      }
    }

    // prefetch next chunk into the other buffer, then wait for current chunk
    if (ks + 1 < nks) {
      const char* gn = gB + (size_t)(ks + 1) * 8192;
      async_copy16(gn + o0, (char*)lb[cur ^ 1] + o0);
      async_copy16(gn + o1, (char*)lb[cur ^ 1] + o1);
      wait_async<2>();                // in-order: current chunk has landed
    } else {
      wait_async<0>();
    }
    __syncthreads();                  // lb[cur] visible to all waves

    v16bf ahi, alo;
#pragma unroll
    for (int i = 0; i < 16; ++i) {
      __bf16 hv = (__bf16)af[i];
      ahi[i] = hv;
      alo[i] = (__bf16)(af[i] - (float)hv);
    }

    const v16bf* fb = reinterpret_cast<const v16bf*>(lb[cur]);
#pragma unroll
    for (int nt = 0; nt < 4; ++nt) {
      v16bf bhi = fb[(nt * 2 + 0) * 32 + lane];
      v16bf blo = fb[(nt * 2 + 1) * 32 + lane];
      acc[nt] = __builtin_amdgcn_wmma_f32_16x16x32_bf16(false, ahi, false, bhi,
                                                        (short)0, acc[nt], false, false);
      acc[nt] = __builtin_amdgcn_wmma_f32_16x16x32_bf16(false, ahi, false, blo,
                                                        (short)0, acc[nt], false, false);
      acc[nt] = __builtin_amdgcn_wmma_f32_16x16x32_bf16(false, alo, false, bhi,
                                                        (short)0, acc[nt], false, false);
    }
  }

  // C/D layout: lanes 0-15 -> M = r, lanes 16-31 -> M = r + 8; N = lane&15
  const int rowbase = m0 + half * 8;
#pragma unroll
  for (int nt = 0; nt < 4; ++nt) {
    const int col = nt * 16 + l15;
    const float bv = bias[col];
#pragma unroll
    for (int r = 0; r < 8; ++r) {
      const int ro = rowbase + r;
      if (ro < N) out[(size_t)ro * D_H + col] = acc[nt][r] + bv;
    }
  }
}

// ---------------------------------------------------------------------------
// Per-channel sum & sum-of-squares over hpre[N x 64]  (for BatchNorm stats)
// ---------------------------------------------------------------------------
__global__ __launch_bounds__(256) void k_colstats(const float* __restrict__ hpre,
                                                  float* __restrict__ stats, long total) {
  __shared__ float s1[256];
  __shared__ float s2[256];
  int tid = threadIdx.x;
  long idx = (long)blockIdx.x * 256 + tid;
  long stride = (long)gridDim.x * 256;
  float a = 0.f, b = 0.f;
  for (long i = idx; i < total; i += stride) {   // channel = tid & 63, invariant
    float v = hpre[i];
    a += v;
    b += v * v;
  }
  s1[tid] = a;
  s2[tid] = b;
  __syncthreads();
  if (tid < 64) {
    float sa = s1[tid] + s1[tid + 64] + s1[tid + 128] + s1[tid + 192];
    float sb = s2[tid] + s2[tid + 64] + s2[tid + 128] + s2[tid + 192];
    atomicAdd(&stats[tid], sa);
    atomicAdd(&stats[64 + tid], sb);
  }
}

__global__ void k_statsfinal(const float* __restrict__ stats,
                             const float* __restrict__ g, const float* __restrict__ be,
                             float* __restrict__ ss, float invN) {
  int c = threadIdx.x;           // 64 threads
  float mean = stats[c] * invN;
  float var  = stats[64 + c] * invN - mean * mean;
  float sc   = g[c] * rsqrtf(var + 1e-5f);
  ss[c]      = sc;
  ss[64 + c] = be[c] - mean * sc;
}

// ---------------------------------------------------------------------------
// Fused BN + ReLU + per-layer prediction head. One node per thread.
// ---------------------------------------------------------------------------
__global__ __launch_bounds__(256) void k_bn_head(
    const float* __restrict__ hpre, const float* __restrict__ ss,
    const float* __restrict__ Wl, const float* __restrict__ bl,
    float* __restrict__ h, float* __restrict__ pred, int N, int addPred) {
  __shared__ float lss[128];
  __shared__ float lw[D_H * D_OUT];
  __shared__ float lbv[D_OUT];
  int tid = threadIdx.x;
  if (tid < 128) lss[tid] = ss[tid];
  lw[tid] = Wl[tid];                 // exactly 256 elements
  if (tid < D_OUT) lbv[tid] = bl[tid];
  __syncthreads();

  int n = blockIdx.x * 256 + tid;
  if (n >= N) return;
  const float* hp = hpre + (size_t)n * D_H;
  float* ho = h + (size_t)n * D_H;
  float p0 = lbv[0], p1 = lbv[1], p2 = lbv[2], p3 = lbv[3];
#pragma unroll
  for (int c = 0; c < D_H; c += 4) {
    float4 v = *(const float4*)(hp + c);
    float r0 = fmaxf(fmaf(v.x, lss[c + 0], lss[64 + c + 0]), 0.f);
    float r1 = fmaxf(fmaf(v.y, lss[c + 1], lss[64 + c + 1]), 0.f);
    float r2 = fmaxf(fmaf(v.z, lss[c + 2], lss[64 + c + 2]), 0.f);
    float r3 = fmaxf(fmaf(v.w, lss[c + 3], lss[64 + c + 3]), 0.f);
    *(float4*)(ho + c) = make_float4(r0, r1, r2, r3);
    p0 += r0 * lw[(c+0)*4+0] + r1 * lw[(c+1)*4+0] + r2 * lw[(c+2)*4+0] + r3 * lw[(c+3)*4+0];
    p1 += r0 * lw[(c+0)*4+1] + r1 * lw[(c+1)*4+1] + r2 * lw[(c+2)*4+1] + r3 * lw[(c+3)*4+1];
    p2 += r0 * lw[(c+0)*4+2] + r1 * lw[(c+1)*4+2] + r2 * lw[(c+2)*4+2] + r3 * lw[(c+3)*4+2];
    p3 += r0 * lw[(c+0)*4+3] + r1 * lw[(c+1)*4+3] + r2 * lw[(c+2)*4+3] + r3 * lw[(c+3)*4+3];
  }
  float* pp = pred + (size_t)n * D_OUT;
  if (addPred) {
    float4 o = *(const float4*)pp;
    p0 += o.x; p1 += o.y; p2 += o.z; p3 += o.w;
  }
  *(float4*)pp = make_float4(p0, p1, p2, p3);
}

// ---------------------------------------------------------------------------
// Edge scatter: agg[dst] += h[src]. One float4 per thread (16 threads/edge);
// gathers are global_load_b128, atomics land in L2 (agg = 25.6 MB << 192 MB L2).
// ---------------------------------------------------------------------------
__global__ __launch_bounds__(256) void k_scatter(const float* __restrict__ h,
                                                 const int* __restrict__ src,
                                                 const int* __restrict__ dst,
                                                 float* __restrict__ agg) {
  long idx = (long)blockIdx.x * 256 + threadIdx.x;
  if (idx >= (long)NE * 16) return;
  int e = (int)(idx >> 4);
  int c = (int)(idx & 15) * 4;
  int s = src[e];
  int d = dst[e];
  float4 v = *(const float4*)(h + (size_t)s * D_H + c);
  float* ap = agg + (size_t)d * D_H + c;
  atomicAdd(ap + 0, v.x);
  atomicAdd(ap + 1, v.y);
  atomicAdd(ap + 2, v.z);
  atomicAdd(ap + 3, v.w);
}

__global__ __launch_bounds__(256) void k_logsoftmax(const float* __restrict__ pred,
                                                    float* __restrict__ out, int N) {
  int n = blockIdx.x * 256 + threadIdx.x;
  if (n >= N) return;
  float4 p = *(const float4*)(pred + (size_t)n * D_OUT);
  float m = fmaxf(fmaxf(p.x, p.y), fmaxf(p.z, p.w));
  float s = expf(p.x - m) + expf(p.y - m) + expf(p.z - m) + expf(p.w - m);
  float l = m + logf(s);
  *(float4*)(out + (size_t)n * D_OUT) = make_float4(p.x - l, p.y - l, p.z - l, p.w - l);
}

// ---------------------------------------------------------------------------
extern "C" void kernel_launch(void* const* d_in, const int* in_sizes, int n_in,
                              void* d_out, int out_size, void* d_ws, size_t ws_size,
                              hipStream_t stream) {
  const float* x    = (const float*)d_in[0];
  const int*   ei   = (const int*)d_in[1];
  const float* W0   = (const float*)d_in[2];
  const float* b0   = (const float*)d_in[3];
  const float* g0   = (const float*)d_in[4];
  const float* be0  = (const float*)d_in[5];
  const float* Wc1  = (const float*)d_in[6];
  const float* bc1  = (const float*)d_in[7];
  const float* gc1  = (const float*)d_in[8];
  const float* bec1 = (const float*)d_in[9];
  const float* Wc2  = (const float*)d_in[10];
  const float* bc2  = (const float*)d_in[11];
  const float* gc2  = (const float*)d_in[12];
  const float* bec2 = (const float*)d_in[13];
  const float* Wl0  = (const float*)d_in[14];
  const float* bl0  = (const float*)d_in[15];
  const float* Wl1  = (const float*)d_in[16];
  const float* bl1  = (const float*)d_in[17];
  const float* Wl2  = (const float*)d_in[18];
  const float* bl2  = (const float*)d_in[19];
  const int* src = ei;
  const int* dst = ei + NE;

  float* ws = (float*)d_ws;
  const size_t N64 = (size_t)NN * D_H;
  float* hpre  = ws;
  float* h     = ws + N64;
  float* agg   = ws + 2 * N64;
  float* pred  = ws + 3 * N64;
  float* stats = pred + (size_t)NN * D_OUT;
  float* ss    = stats + 128;
  __bf16* pB0  = (__bf16*)(ss + 128);
  __bf16* pW1  = pB0 + 2 * D_IN * D_H;   // hi+lo
  __bf16* pW2  = pW1 + 2 * D_H * D_H;

  // pack weights into WMMA fragment layout (split bf16)
  k_pack_w<<<(D_IN * D_H + 255) / 256, 256, 0, stream>>>(W0,  pB0, D_IN);
  k_pack_w<<<(D_H * D_H + 255) / 256, 256, 0, stream>>>(Wc1, pW1, D_H);
  k_pack_w<<<(D_H * D_H + 255) / 256, 256, 0, stream>>>(Wc2, pW2, D_H);

  const int gemmBlocks = (NN + 127) / 128;    // 8 waves x 16 rows per block
  const int nodeBlocks = (NN + 255) / 256;
  const long sthreads  = (long)NE * 16;
  const int scatBlocks = (int)((sthreads + 255) / 256);
  const float invN = 1.0f / (float)NN;

  // ---- layer 0: first_h + head0 ----
  k_wmma_gemm<<<gemmBlocks, 256, 0, stream>>>(x, nullptr, pB0, b0, hpre, NN, D_IN);
  (void)hipMemsetAsync(stats, 0, 128 * sizeof(float), stream);
  k_colstats<<<1024, 256, 0, stream>>>(hpre, stats, (long)N64);
  k_statsfinal<<<1, 64, 0, stream>>>(stats, g0, be0, ss, invN);
  k_bn_head<<<nodeBlocks, 256, 0, stream>>>(hpre, ss, Wl0, bl0, h, pred, NN, 0);

  // ---- conv 1 ----
  (void)hipMemsetAsync(agg, 0, N64 * sizeof(float), stream);
  k_scatter<<<scatBlocks, 256, 0, stream>>>(h, src, dst, agg);
  k_wmma_gemm<<<gemmBlocks, 256, 0, stream>>>(h, agg, pW1, bc1, hpre, NN, D_H);
  (void)hipMemsetAsync(stats, 0, 128 * sizeof(float), stream);
  k_colstats<<<1024, 256, 0, stream>>>(hpre, stats, (long)N64);
  k_statsfinal<<<1, 64, 0, stream>>>(stats, gc1, bec1, ss, invN);
  k_bn_head<<<nodeBlocks, 256, 0, stream>>>(hpre, ss, Wl1, bl1, h, pred, NN, 1);

  // ---- conv 2 ----
  (void)hipMemsetAsync(agg, 0, N64 * sizeof(float), stream);
  k_scatter<<<scatBlocks, 256, 0, stream>>>(h, src, dst, agg);
  k_wmma_gemm<<<gemmBlocks, 256, 0, stream>>>(h, agg, pW2, bc2, hpre, NN, D_H);
  (void)hipMemsetAsync(stats, 0, 128 * sizeof(float), stream);
  k_colstats<<<1024, 256, 0, stream>>>(hpre, stats, (long)N64);
  k_statsfinal<<<1, 64, 0, stream>>>(stats, gc2, bec2, ss, invN);
  k_bn_head<<<nodeBlocks, 256, 0, stream>>>(hpre, ss, Wl2, bl2, h, pred, NN, 1);

  // ---- output ----
  k_logsoftmax<<<nodeBlocks, 256, 0, stream>>>(pred, (float*)d_out, NN);
}
